// FFJORD_58798102282496
// MI455X (gfx1250) — matrix operand order repaired and checked by reading, not compiled, not run on hardware
//
#include <hip/hip_runtime.h>

typedef __bf16 bf16;
typedef __attribute__((ext_vector_type(16))) __bf16 v16bf;
typedef __attribute__((ext_vector_type(8)))  __bf16 v8bf;
typedef __attribute__((ext_vector_type(8)))  float  v8f;

#define BATCH 8192
#define FEA   64
#define HID   512
#define KA    96      // padded augmented-K (65 -> 96, zeros)
#define BM    32      // batch rows per workgroup
#define NTHR  256
#define NWAVE 8

// ---------------- fast tanh (gfx1250 v_tanh_f32) ----------------

__device__ __forceinline__ float fast_tanh(float x) {
#if __has_builtin(__builtin_amdgcn_tanhf)
  return __builtin_amdgcn_tanhf(x);
#else
  return tanhf(x);
#endif
}

// ---------------- WMMA helpers (gfx1250, wave32) ----------------

__device__ __forceinline__ v8f wmma_bf16(v16bf a, v16bf b, v8f c) {
  // D = A(16x32 bf16) x B(32x16 bf16) + C(16x16 f32)
  return __builtin_amdgcn_wmma_f32_16x16x32_bf16(false, a, false, b,
                                                 (short)0, c, false, false);
}

// A fragment: row-major [16 x >=32] bf16 tile starting at p, row stride `stride`.
// ISA layout: lane<16 -> row=lane, K 0..7 & 16..23 ; lane>=16 -> same rows, K 8..15 & 24..31.
__device__ __forceinline__ v16bf load_frag_a(const bf16* p, int stride, int lane) {
  const int m = lane & 15, hi = lane >> 4;
  const bf16* r = p + m * stride + hi * 8;
  v8bf x0 = *(const v8bf*)(r);
  v8bf x1 = *(const v8bf*)(r + 16);
  return __builtin_shufflevector(x0, x1, 0, 1, 2, 3, 4, 5, 6, 7,
                                 8, 9, 10, 11, 12, 13, 14, 15);
}

// B fragment from TRANSPOSED storage Bt[N][K] (row n holds K consecutive).
// ISA layout: lane<16 -> col n=lane, K 0..7 & 16..23 ; lane>=16 -> K 8..15 & 24..31.
__device__ __forceinline__ v16bf load_frag_bt(const bf16* p, int stride, int lane) {
  const int n = lane & 15, hi = lane >> 4;
  const bf16* r = p + n * stride + hi * 8;
  v8bf x0 = *(const v8bf*)(r);
  v8bf x1 = *(const v8bf*)(r + 16);
  return __builtin_shufflevector(x0, x1, 0, 1, 2, 3, 4, 5, 6, 7,
                                 8, 9, 10, 11, 12, 13, 14, 15);
}

// One 16x16 output tile, K-split across NACC independent accumulators so the
// WMMA RAW hazard slots are filled with independent WMMAs instead of v_nops.
template <int K, int AS, int BS, int NACC>
__device__ __forceinline__ v8f mm_tile(const bf16* Ap, const bf16* Bp, int lane) {
  static_assert(K % (32 * NACC) == 0, "K must divide into NACC chains");
  const v8f vzero = {0.f, 0.f, 0.f, 0.f, 0.f, 0.f, 0.f, 0.f};
  v8f acc[NACC];
#pragma unroll
  for (int u = 0; u < NACC; ++u) acc[u] = vzero;
#pragma unroll
  for (int k0 = 0; k0 < K; k0 += 32 * NACC) {
#pragma unroll
    for (int u = 0; u < NACC; ++u) {
      const int kk = k0 + u * 32;
      v16bf a = load_frag_a(Ap + kk, AS, lane);
      v16bf b = load_frag_bt(Bp + kk, BS, lane);
      acc[u] = wmma_bf16(a, b, acc[u]);
    }
  }
#pragma unroll
  for (int u = 1; u < NACC; ++u)
#pragma unroll
    for (int j = 0; j < 8; ++j) acc[0][j] += acc[u][j];
  return acc[0];
}

// ---------------- weight prep: bf16 convert + transpose ----------------

__global__ void ffjord_prep(const float* __restrict__ W1, const float* __restrict__ W2,
                            const float* __restrict__ W3,
                            bf16* __restrict__ W1t, bf16* __restrict__ W2t,
                            bf16* __restrict__ W3t, bf16* __restrict__ W2b,
                            bf16* __restrict__ W3b) {
  const int gsz = gridDim.x * blockDim.x;
  const int g = blockIdx.x * blockDim.x + threadIdx.x;
  // W1t[n][k] = W1[k][n] (k<65 else 0), shape [HID][KA]
  for (int i = g; i < HID * KA; i += gsz) {
    int n = i / KA, k = i - n * KA;
    float v = (k < FEA + 1) ? W1[k * HID + n] : 0.0f;
    W1t[i] = (bf16)v;
  }
  // W2t[n][k] = W2[k][n]; W2b = bf16(W2) row-major
  for (int i = g; i < HID * HID; i += gsz) {
    int n = i / HID, k = i - n * HID;
    W2t[i] = (bf16)W2[k * HID + n];
    W2b[i] = (bf16)W2[i];
  }
  // W3 is [HID][FEA]. W3t[n][k] = W3[k][n], shape [FEA][HID]
  for (int i = g; i < FEA * HID; i += gsz) {
    int n = i / HID, k = i - n * HID;
    W3t[i] = (bf16)W3[k * FEA + n];
  }
  // W3b = bf16(W3) row-major [HID][FEA]
  for (int i = g; i < HID * FEA; i += gsz) {
    W3b[i] = (bf16)W3[i];
  }
}

// ---------------- fused FFJORD RK4 kernel ----------------

__global__ __launch_bounds__(NTHR, 1) void ffjord_kernel(
    const float* __restrict__ x, const float* __restrict__ eps,
    const float* __restrict__ b1, const float* __restrict__ b2,
    const float* __restrict__ b3,
    const bf16* __restrict__ W1t, const bf16* __restrict__ W2t,
    const bf16* __restrict__ W3t, const bf16* __restrict__ W2b,
    const bf16* __restrict__ W3b,
    const int* __restrict__ nsteps_p, float* __restrict__ out) {
  __shared__ bf16 sXa[BM * KA];     // augmented input tile (bf16)
  __shared__ bf16 sH1[BM * HID];    // tanh(u1); later a2-mask source
  __shared__ bf16 sH2[BM * HID];    // tanh(u2); later masked a3
  __shared__ bf16 sWe[BM * HID];    // eps @ W1[0:64,:]
  __shared__ bf16 sEps[BM * FEA];
  __shared__ float sX[BM * FEA];    // current state
  __shared__ float sDx[BM * FEA];   // dynamics output (k_i)
  __shared__ float sAcc[BM * FEA];  // RK accumulator
  __shared__ float sTr[BM];
  __shared__ float sLd[BM];
  __shared__ float sLacc[BM];

  const int tid = threadIdx.x;
  const int lane = tid & 31;
  const int w = tid >> 5;
  const int row0 = blockIdx.x * BM;
  const int fn = lane & 15;          // fragment N index
  const int fm = (lane >> 4) << 3;   // fragment M base (0 or 8)

  // ---- init state / eps tiles ----
  for (int i = tid; i < BM * FEA; i += NTHR) {
    int m = i / FEA, f = i - m * FEA;
    sX[i] = x[(row0 + m) * FEA + f];
    sDx[i] = 0.0f;
    sEps[i] = (bf16)eps[(row0 + m) * FEA + f];
  }
  for (int i = tid; i < BM; i += NTHR) sLd[i] = 0.0f;
  __syncthreads();

  // ---- one-time: Weps = eps @ W1[0:64,:]  ([BM,512]) ----
#pragma unroll 1
  for (int pp = 0; pp < (BM / 16) * (HID / 16) / NWAVE; ++pp) {
    const int tt = w + pp * NWAVE;
    const int m0 = (tt & 1) * 16, n0 = (tt >> 1) * 16;
    v8f c = mm_tile<FEA, FEA, KA, 2>(sEps + m0 * FEA, W1t + n0 * KA, lane);
    const int n = n0 + fn, mb = m0 + fm;
#pragma unroll
    for (int j = 0; j < 8; ++j) sWe[(mb + j) * HID + n] = (bf16)c[j];
  }

  const int ns = nsteps_p[0];
  const int nint = (ns > 1) ? (ns - 1) : 1;
  const float hstep = 1.0f / (float)nint;

  for (int s = 0; s < nint; ++s) {
    const float t0 = hstep * (float)s;
    __syncthreads();
    for (int i = tid; i < BM * FEA; i += NTHR) sAcc[i] = 0.0f;
    for (int i = tid; i < BM; i += NTHR) sLacc[i] = 0.0f;

    for (int st = 0; st < 4; ++st) {
      const float coef = (st == 0) ? 0.0f : ((st == 3) ? hstep : 0.5f * hstep);
      const float tt_ = t0 + coef;
      const float wgt = (st == 1 || st == 2) ? 2.0f : 1.0f;
      __syncthreads();
      // build augmented bf16 input: [x + coef*k_prev | t | 0-pad]
      for (int i = tid; i < BM * KA; i += NTHR) {
        int m = i / KA, k = i - m * KA;
        float v = (k < FEA) ? (sX[m * FEA + k] + coef * sDx[m * FEA + k])
                            : ((k == FEA) ? tt_ : 0.0f);
        sXa[i] = (bf16)v;
      }
      for (int i = tid; i < BM; i += NTHR) sTr[i] = 0.0f;
      __syncthreads();

      // GEMM1: H1 = tanh(Xa @ W1 + b1)   [BM,512], K=96
#pragma unroll 1
      for (int pp = 0; pp < (BM / 16) * (HID / 16) / NWAVE; ++pp) {
        const int tt = w + pp * NWAVE;
        const int m0 = (tt & 1) * 16, n0 = (tt >> 1) * 16;
        v8f c = mm_tile<KA, KA, KA, 3>(sXa + m0 * KA, W1t + n0 * KA, lane);
        const int n = n0 + fn, mb = m0 + fm;
        const float bb = b1[n];
#pragma unroll
        for (int j = 0; j < 8; ++j)
          sH1[(mb + j) * HID + n] = (bf16)fast_tanh(c[j] + bb);
      }
      __syncthreads();

      // GEMM2: H2 = tanh(H1 @ W2 + b2)   [BM,512], K=512
#pragma unroll 1
      for (int pp = 0; pp < (BM / 16) * (HID / 16) / NWAVE; ++pp) {
        const int tt = w + pp * NWAVE;
        const int m0 = (tt & 1) * 16, n0 = (tt >> 1) * 16;
        v8f c = mm_tile<HID, HID, HID, 4>(sH1 + m0 * HID, W2t + n0 * HID, lane);
        const int n = n0 + fn, mb = m0 + fm;
        const float bb = b2[n];
#pragma unroll
        for (int j = 0; j < 8; ++j)
          sH2[(mb + j) * HID + n] = (bf16)fast_tanh(c[j] + bb);
      }
      __syncthreads();

      // GEMM3: dx = H2 @ W3 + b3   [BM,64], K=512  (8 tiles, 1 per wave)
      {
        const int m0 = (w & 1) * 16, n0 = (w >> 1) * 16;
        v8f c = mm_tile<HID, HID, HID, 4>(sH2 + m0 * HID, W3t + n0 * HID, lane);
        const int n = n0 + fn, mb = m0 + fm;
        const float bb = b3[n];
#pragma unroll
        for (int j = 0; j < 8; ++j)
          sDx[(mb + j) * FEA + n] = c[j] + bb;
      }
      __syncthreads();

      // VJP-1: a3 = eps @ W3^T, masked by (1-h2^2), stored into sH2
#pragma unroll 1
      for (int pp = 0; pp < (BM / 16) * (HID / 16) / NWAVE; ++pp) {
        const int tt = w + pp * NWAVE;
        const int m0 = (tt & 1) * 16, n0 = (tt >> 1) * 16;
        v8f c = mm_tile<FEA, FEA, FEA, 2>(sEps + m0 * FEA, W3b + n0 * FEA, lane);
        const int n = n0 + fn, mb = m0 + fm;
#pragma unroll
        for (int j = 0; j < 8; ++j) {
          const int idx = (mb + j) * HID + n;
          const float h2v = (float)sH2[idx];
          sH2[idx] = (bf16)(c[j] * (1.0f - h2v * h2v));
        }
      }
      __syncthreads();

      // VJP-2: a2 = a3m @ W2^T, mask (1-h1^2), tr += a2m .* Weps (row reduce)
#pragma unroll 1
      for (int pp = 0; pp < (BM / 16) * (HID / 16) / NWAVE; ++pp) {
        const int tt = w + pp * NWAVE;
        const int m0 = (tt & 1) * 16, n0 = (tt >> 1) * 16;
        v8f c = mm_tile<HID, HID, HID, 4>(sH2 + m0 * HID, W2b + n0 * HID, lane);
        const int n = n0 + fn, mb = m0 + fm;
#pragma unroll
        for (int j = 0; j < 8; ++j) {
          const int idx = (mb + j) * HID + n;
          const float h1v = (float)sH1[idx];
          const float a2m = c[j] * (1.0f - h1v * h1v);
          float p = a2m * (float)sWe[idx];
          // reduce across the 16-lane N group
          p += __shfl_xor(p, 1, 32);
          p += __shfl_xor(p, 2, 32);
          p += __shfl_xor(p, 4, 32);
          p += __shfl_xor(p, 8, 32);
          if ((lane & 15) == 0) atomicAdd(&sTr[mb + j], p);
        }
      }
      __syncthreads();

      // RK accumulate
      for (int i = tid; i < BM * FEA; i += NTHR) sAcc[i] += wgt * sDx[i];
      for (int i = tid; i < BM; i += NTHR) sLacc[i] += wgt * sTr[i];
    }  // stages

    __syncthreads();
    const float scl = hstep / 6.0f;
    for (int i = tid; i < BM * FEA; i += NTHR) sX[i] += scl * sAcc[i];
    for (int i = tid; i < BM; i += NTHR) sLd[i] -= scl * sLacc[i];
    __syncthreads();
  }  // time steps

  // ---- write outputs: z then log_det ----
  for (int i = tid; i < BM * FEA; i += NTHR) {
    int m = i / FEA, f = i - m * FEA;
    out[(row0 + m) * FEA + f] = sX[i];
  }
  float* out_ld = out + BATCH * FEA;
  for (int i = tid; i < BM; i += NTHR) out_ld[row0 + i] = sLd[i];
}

// ---------------- host launch ----------------

extern "C" void kernel_launch(void* const* d_in, const int* in_sizes, int n_in,
                              void* d_out, int out_size, void* d_ws, size_t ws_size,
                              hipStream_t stream) {
  (void)in_sizes; (void)n_in; (void)out_size; (void)ws_size;
  const float* x   = (const float*)d_in[0];
  const float* eps = (const float*)d_in[1];
  const float* W1  = (const float*)d_in[2];
  const float* b1  = (const float*)d_in[3];
  const float* W2  = (const float*)d_in[4];
  const float* b2  = (const float*)d_in[5];
  const float* W3  = (const float*)d_in[6];
  const float* b3  = (const float*)d_in[7];
  const int*   nst = (const int*)d_in[8];
  float* out = (float*)d_out;

  char* ws = (char*)d_ws;
  size_t off = 0;
  bf16* W1t = (bf16*)(ws + off); off += (size_t)HID * KA * 2;    // 98304
  bf16* W2t = (bf16*)(ws + off); off += (size_t)HID * HID * 2;   // 524288
  bf16* W3t = (bf16*)(ws + off); off += (size_t)FEA * HID * 2;   // 65536
  bf16* W2b = (bf16*)(ws + off); off += (size_t)HID * HID * 2;   // 524288
  bf16* W3b = (bf16*)(ws + off); off += (size_t)HID * FEA * 2;   // 65536

  hipLaunchKernelGGL(ffjord_prep, dim3(512), dim3(256), 0, stream,
                     W1, W2, W3, W1t, W2t, W3t, W2b, W3b);
  hipLaunchKernelGGL(ffjord_kernel, dim3(BATCH / BM), dim3(NTHR), 0, stream,
                     x, eps, b1, b2, b3, W1t, W2t, W3t, W2b, W3b, nst, out);
}
